// QuantizedConv1D_29884382445886
// MI455X (gfx1250) — matrix operand order repaired and checked by reading, not compiled, not run on hardware
//
#include <hip/hip_runtime.h>
#include <hip/hip_bf16.h>

typedef _Float16 v4h  __attribute__((ext_vector_type(4)));
typedef _Float16 v8h  __attribute__((ext_vector_type(8)));
typedef _Float16 v16h __attribute__((ext_vector_type(16)));
typedef float    v4f  __attribute__((ext_vector_type(4)));
typedef float    v8f  __attribute__((ext_vector_type(8)));
typedef int      v4i  __attribute__((ext_vector_type(4)));

#define GK     1024      // NX (K)
#define GN     4096      // NF (N)
#define TILE_M 128
#define TILE_N 128
#define TILE_K 32
#define LDSS   40        // padded LDS row stride in halves (80 B)

static __device__ __forceinline__ v16h cat8(v8h a, v8h b) {
    return __builtin_shufflevector(a, b, 0,1,2,3,4,5,6,7,8,9,10,11,12,13,14,15);
}

// ---------------- async copy primitives (CDNA5) ----------------
typedef __attribute__((address_space(1))) v4i* g_v4i_ptr;
typedef __attribute__((address_space(3))) v4i* l_v4i_ptr;

static __device__ __forceinline__ void async_copy_b128(const _Float16* gsrc, unsigned lds_off) {
#if __has_builtin(__builtin_amdgcn_global_load_async_to_lds_b128)
    __builtin_amdgcn_global_load_async_to_lds_b128(
        (g_v4i_ptr)(unsigned long long)gsrc,
        (l_v4i_ptr)(unsigned long long)lds_off,
        0, 0);
#else
    asm volatile("global_load_async_to_lds_b128 %0, %1, off"
                 :: "v"(lds_off), "v"(gsrc) : "memory");
#endif
}
static __device__ __forceinline__ void wait_async_le6() {
#if __has_builtin(__builtin_amdgcn_s_wait_asynccnt)
    __builtin_amdgcn_s_wait_asynccnt(6);
#else
    asm volatile("s_wait_asynccnt 0x6" ::: "memory");
#endif
}
static __device__ __forceinline__ void wait_async_0() {
#if __has_builtin(__builtin_amdgcn_s_wait_asynccnt)
    __builtin_amdgcn_s_wait_asynccnt(0);
#else
    asm volatile("s_wait_asynccnt 0x0" ::: "memory");
#endif
}

// ---------------- phase A: one-time conversions ----------------
// X fp32 [M,K] -> Xhi,Xlo f16 [M,K]  (two-term split, ~22 mantissa bits total)
__global__ __launch_bounds__(256)
void cvt_x_split(const float* __restrict__ X, _Float16* __restrict__ Xhi,
                 _Float16* __restrict__ Xlo, int n4) {
    int i = blockIdx.x * 256 + threadIdx.x;
    if (i >= n4) return;
    v4f x = ((const v4f*)X)[i];
    v4h h, l;
#pragma unroll
    for (int j = 0; j < 4; ++j) {
        _Float16 fh = (_Float16)x[j];
        h[j] = fh;
        l[j] = (_Float16)(x[j] - (float)fh);
    }
    ((v4h*)Xhi)[i] = h;
    ((v4h*)Xlo)[i] = l;
}

// W fp32 [K,N] -> Wt f16 [N,K] (transpose; int8-range values are exact in f16)
__global__ __launch_bounds__(256)
void cvt_w_transpose(const float* __restrict__ W, _Float16* __restrict__ Wt) {
    __shared__ __align__(16) _Float16 sT[64 * 72];   // [n_local][k_local], stride 72
    const int tid = threadIdx.x;
    const int n0 = blockIdx.x * 64;
    const int k0 = blockIdx.y * 64;
    // read 64(K) x 64(N) fp32 tile, convert, scatter transposed into LDS
    const int c4 = tid & 15;        // float4 col in N
    const int r0 = tid >> 4;        // K row (0..15), +16*i
#pragma unroll
    for (int i = 0; i < 4; ++i) {
        const int r = r0 + 16 * i;
        v4f w = *(const v4f*)(W + (size_t)(k0 + r) * GN + n0 + c4 * 4);
#pragma unroll
        for (int j = 0; j < 4; ++j)
            sT[(c4 * 4 + j) * 72 + r] = (_Float16)w[j];
    }
    __syncthreads();
    // write coalesced b128 rows of Wt
    const int cc = tid & 7;         // v8h col in K
    const int rr0 = tid >> 3;       // N row (0..31), +32*i
#pragma unroll
    for (int i = 0; i < 2; ++i) {
        const int rr = rr0 + 32 * i;
        v8h v = *(const v8h*)&sT[rr * 72 + cc * 8];
        *(v8h*)(Wt + (size_t)(n0 + rr) * GK + k0 + cc * 8) = v;
    }
}

// ---------------- phase B: async double-buffered WMMA GEMM ----------------
__global__ __launch_bounds__(256)
void qgemm_wmma_async(const _Float16* __restrict__ Xhi, const _Float16* __restrict__ Xlo,
                      const _Float16* __restrict__ Wt,
                      const float* __restrict__ scales, const float* __restrict__ bias,
                      float* __restrict__ Out)
{
    __shared__ __align__(16) _Float16 sAhi[2][TILE_M * LDSS];
    __shared__ __align__(16) _Float16 sAlo[2][TILE_M * LDSS];
    __shared__ __align__(16) _Float16 sBt [2][TILE_N * LDSS];

    const int tid  = threadIdx.x;
    const int lane = tid & 31;
    const int wave = tid >> 5;
    const int half = lane >> 4;
    const int l16  = lane & 15;

    const int tileN0 = blockIdx.x * TILE_N;
    const int tileM0 = blockIdx.y * TILE_M;
    const int waveM = wave >> 2;   // 0..1 -> 64 rows
    const int waveN = wave & 3;    // 0..3 -> 32 cols

    const unsigned aHiB[2] = { (unsigned)(unsigned long long)&sAhi[0][0],
                               (unsigned)(unsigned long long)&sAhi[1][0] };
    const unsigned aLoB[2] = { (unsigned)(unsigned long long)&sAlo[0][0],
                               (unsigned)(unsigned long long)&sAlo[1][0] };
    const unsigned bB[2]   = { (unsigned)(unsigned long long)&sBt[0][0],
                               (unsigned)(unsigned long long)&sBt[1][0] };

    v8f acc[4][2];
#pragma unroll
    for (int mt = 0; mt < 4; ++mt)
#pragma unroll
        for (int nt = 0; nt < 2; ++nt)
            acc[mt][nt] = (v8f){0.f,0.f,0.f,0.f,0.f,0.f,0.f,0.f};

    // per-lane copy mapping: 512 16B chunks per 128x32 f16 tile, 2 per lane
    auto issue_tile = [&](int buf, int k0) {
#pragma unroll
        for (int i = 0; i < 2; ++i) {
            const int c = tid + 256 * i;
            const int r = c >> 2;          // tile row
            const int q = c & 3;           // 16B chunk within row
            const unsigned loff = (unsigned)((r * LDSS + q * 8) * 2);
            const size_t gAoff = (size_t)(tileM0 + r) * GK + k0 + q * 8;
            const size_t gBoff = (size_t)(tileN0 + r) * GK + k0 + q * 8;
            async_copy_b128(Xhi + gAoff, aHiB[buf] + loff);
            async_copy_b128(Xlo + gAoff, aLoB[buf] + loff);
            async_copy_b128(Wt  + gBoff, bB[buf]   + loff);
        }
    };

    issue_tile(0, 0);
    int buf = 0;
    for (int k0 = 0; k0 < GK; k0 += TILE_K) {
        if (k0 + TILE_K < GK) {
            issue_tile(buf ^ 1, k0 + TILE_K);   // prefetch next tile into other buffer
            wait_async_le6();                   // retire current tile (in-order completion)
        } else {
            wait_async_0();
        }
        __syncthreads();

        // B fragments: lane holds column n, K at 8*half+{0..7} and 16+8*half+{0..7}
        v16h bfrag[2];
#pragma unroll
        for (int nt = 0; nt < 2; ++nt) {
            const _Float16* p = &sBt[buf][(waveN*32 + nt*16 + l16) * LDSS + half*8];
            bfrag[nt] = cat8(*(const v8h*)p, *(const v8h*)(p + 16));
        }
        // A fragments + WMMA (hi/lo split accumulate)
#pragma unroll
        for (int mt = 0; mt < 4; ++mt) {
            const int m = waveM*64 + mt*16 + l16;
            const _Float16* ph = &sAhi[buf][m * LDSS + half*8];
            const _Float16* pl = &sAlo[buf][m * LDSS + half*8];
            v16h ahi = cat8(*(const v8h*)ph, *(const v8h*)(ph + 16));
            v16h alo = cat8(*(const v8h*)pl, *(const v8h*)(pl + 16));
#pragma unroll
            for (int nt = 0; nt < 2; ++nt) {
                acc[mt][nt] = __builtin_amdgcn_wmma_f32_16x16x32_f16(
                    false, alo, false, bfrag[nt], (short)0, acc[mt][nt], false, false);
                acc[mt][nt] = __builtin_amdgcn_wmma_f32_16x16x32_f16(
                    false, ahi, false, bfrag[nt], (short)0, acc[mt][nt], false, false);
            }
        }
        __syncthreads();   // all waves done with buf before it is overwritten
        buf ^= 1;
    }

    // epilogue: per-channel scale + bias
#pragma unroll
    for (int nt = 0; nt < 2; ++nt) {
        const int n = tileN0 + waveN*32 + nt*16 + l16;
        const float sc = scales[n];
        const float bs = bias[n];
#pragma unroll
        for (int mt = 0; mt < 4; ++mt) {
            const int mBase = tileM0 + waveM*64 + mt*16 + half*8;
#pragma unroll
            for (int v = 0; v < 8; ++v)
                Out[(size_t)(mBase + v) * GN + n] = acc[mt][nt][v] * sc + bs;
        }
    }
}

// ---------------- fallback: fused single-kernel version (round 1) ----------------
__global__ __launch_bounds__(256)
void qlinear_wmma_fused(const float* __restrict__ X, const float* __restrict__ W,
                        const float* __restrict__ scales, const float* __restrict__ bias,
                        float* __restrict__ Out)
{
    __shared__ __align__(16) _Float16 sAhi[TILE_M * LDSS];
    __shared__ __align__(16) _Float16 sAlo[TILE_M * LDSS];
    __shared__ __align__(16) _Float16 sBt [TILE_N * LDSS];

    const int tid  = threadIdx.x;
    const int lane = tid & 31;
    const int wave = tid >> 5;
    const int half = lane >> 4;
    const int l16  = lane & 15;
    const int tileN0 = blockIdx.x * TILE_N;
    const int tileM0 = blockIdx.y * TILE_M;
    const int waveM = wave >> 2;
    const int waveN = wave & 3;

    v8f acc[4][2];
#pragma unroll
    for (int mt = 0; mt < 4; ++mt)
#pragma unroll
        for (int nt = 0; nt < 2; ++nt)
            acc[mt][nt] = (v8f){0.f,0.f,0.f,0.f,0.f,0.f,0.f,0.f};

    const int aRow  = tid >> 3;
    const int aCol4 = tid & 7;
    const int bRow  = tid >> 5;
    const int bCol4 = tid & 31;

    for (int k0 = 0; k0 < GK; k0 += TILE_K) {
        v4f aReg[4], bReg[4];
#pragma unroll
        for (int i = 0; i < 4; ++i) {
            aReg[i] = *(const v4f*)(X + (size_t)(tileM0 + aRow + 32*i) * GK + k0 + aCol4*4);
            bReg[i] = *(const v4f*)(W + (size_t)(k0 + bRow + 8*i) * GN + tileN0 + bCol4*4);
        }
        __syncthreads();
#pragma unroll
        for (int i = 0; i < 4; ++i) {
            const int r = aRow + 32*i;
            v4h h, l;
#pragma unroll
            for (int j = 0; j < 4; ++j) {
                float f = aReg[i][j];
                _Float16 fh = (_Float16)f;
                h[j] = fh;
                l[j] = (_Float16)(f - (float)fh);
            }
            *(v4h*)&sAhi[r * LDSS + aCol4*4] = h;
            *(v4h*)&sAlo[r * LDSS + aCol4*4] = l;
            const int kr = bRow + 8*i;
#pragma unroll
            for (int j = 0; j < 4; ++j)
                sBt[(bCol4*4 + j) * LDSS + kr] = (_Float16)bReg[i][j];
        }
        __syncthreads();
        v16h bfrag[2];
#pragma unroll
        for (int nt = 0; nt < 2; ++nt) {
            const _Float16* p = &sBt[(waveN*32 + nt*16 + l16) * LDSS + half*8];
            bfrag[nt] = cat8(*(const v8h*)p, *(const v8h*)(p + 16));
        }
#pragma unroll
        for (int mt = 0; mt < 4; ++mt) {
            const int m = waveM*64 + mt*16 + l16;
            const _Float16* ph = &sAhi[m * LDSS + half*8];
            const _Float16* pl = &sAlo[m * LDSS + half*8];
            v16h ahi = cat8(*(const v8h*)ph, *(const v8h*)(ph + 16));
            v16h alo = cat8(*(const v8h*)pl, *(const v8h*)(pl + 16));
#pragma unroll
            for (int nt = 0; nt < 2; ++nt) {
                acc[mt][nt] = __builtin_amdgcn_wmma_f32_16x16x32_f16(
                    false, alo, false, bfrag[nt], (short)0, acc[mt][nt], false, false);
                acc[mt][nt] = __builtin_amdgcn_wmma_f32_16x16x32_f16(
                    false, ahi, false, bfrag[nt], (short)0, acc[mt][nt], false, false);
            }
        }
    }
#pragma unroll
    for (int nt = 0; nt < 2; ++nt) {
        const int n = tileN0 + waveN*32 + nt*16 + l16;
        const float sc = scales[n];
        const float bs = bias[n];
#pragma unroll
        for (int mt = 0; mt < 4; ++mt) {
            const int mBase = tileM0 + waveM*64 + mt*16 + half*8;
#pragma unroll
            for (int v = 0; v < 8; ++v)
                Out[(size_t)(mBase + v) * GN + n] = acc[mt][nt][v] * sc + bs;
        }
    }
}

extern "C" void kernel_launch(void* const* d_in, const int* in_sizes, int n_in,
                              void* d_out, int out_size, void* d_ws, size_t ws_size,
                              hipStream_t stream) {
    const float* X  = (const float*)d_in[0];   // [M, 1024]
    const float* W  = (const float*)d_in[1];   // [1024, 4096]
    const float* Sc = (const float*)d_in[2];   // [4096]
    const float* Bi = (const float*)d_in[3];   // [4096]
    float* Out = (float*)d_out;

    const int M = in_sizes[0] / GK;            // 8192
    const size_t xHalves = (size_t)M * GK;
    const size_t need = xHalves * 2 * sizeof(_Float16) + (size_t)GN * GK * sizeof(_Float16);

    dim3 grid(GN / TILE_N, M / TILE_M);

    if (ws_size >= need) {
        _Float16* Xhi = (_Float16*)d_ws;
        _Float16* Xlo = Xhi + xHalves;
        _Float16* Wt  = Xlo + xHalves;
        const int n4 = (int)(xHalves / 4);
        cvt_x_split<<<(n4 + 255) / 256, 256, 0, stream>>>(X, Xhi, Xlo, n4);
        cvt_w_transpose<<<dim3(GN / 64, GK / 64), 256, 0, stream>>>(W, Wt);
        qgemm_wmma_async<<<grid, 256, 0, stream>>>(Xhi, Xlo, Wt, Sc, Bi, Out);
    } else {
        qlinear_wmma_fused<<<grid, 256, 0, stream>>>(X, W, Sc, Bi, Out);
    }
}
